// SideAdapter_21337397526852
// MI455X (gfx1250) — compile-verified, matrix operand order checked by reading, not run on hardware
//
#include <hip/hip_runtime.h>
#include <hip/hip_bf16.h>

typedef __attribute__((ext_vector_type(16))) _Float16 v16h;
typedef __attribute__((ext_vector_type(8)))  float    v8f;

namespace cfg {
constexpr int T = 16, C = 768, NH = 12, Q = 100, GH = 14, IMG = 224;
constexpr int DOUT = 512, NTXT = 32, NL = 3;
constexpr int HD = C / NH;      // 64
constexpr int L  = GH * GH;     // 196
constexpr int S  = Q + 1 + L;   // 297
constexpr int SP = 320;         // padded sequence (20 x 16 tiles)
constexpr int M  = T * SP;      // 5120 GEMM rows
constexpr float SCALE = 0.125f; // 1/sqrt(HD)
}

// ---------------------------------------------------------------------------
// Mask pooling: pooled[t,q,l] = 0 if any mask value in the 16x16 patch >= 0,
// else -1e30 (== -inf in the reference). One 256-thread block per patch.
// ---------------------------------------------------------------------------
__global__ void pool_bias_kernel(const float* __restrict__ masks,
                                 float* __restrict__ pooled) {
  using namespace cfg;
  int pid = blockIdx.x;                 // t*Q*L + q*L + l
  int l = pid % L;
  int q = (pid / L) % Q;
  int t = pid / (L * Q);
  int gr = l / GH, gc = l % GH;
  int i = threadIdx.x / 16, j = threadIdx.x % 16;
  const float* p = masks + (((size_t)t * Q + q) * IMG + gr * 16 + i) * IMG + gc * 16 + j;
  __shared__ int anyv;
  if (threadIdx.x == 0) anyv = 0;
  __syncthreads();
  if (*p >= 0.0f) atomicOr(&anyv, 1);
  __syncthreads();
  if (threadIdx.x == 0) pooled[pid] = anyv ? 0.0f : -1e30f;
}

// ---------------------------------------------------------------------------
// Assemble x [T, SP, C]: rows 0..Q-1 and row Q are cls_token,
// rows Q+1..Q+L are pix_feat, rows >= S zero.
// ---------------------------------------------------------------------------
__global__ void assemble_x_kernel(const float* __restrict__ cls,
                                  const float* __restrict__ pix,
                                  float* __restrict__ x) {
  using namespace cfg;
  int idx = blockIdx.x * 256 + threadIdx.x;
  if (idx >= M * C) return;
  int c = idx % C;
  int s = (idx / C) % SP;
  int t = idx / (C * SP);
  float v;
  if (s >= S)      v = 0.0f;
  else if (s <= Q) v = cls[t * C + c];
  else             v = pix[((size_t)t * C + c) * L + (s - Q - 1)];
  x[idx] = v;
}

__global__ void cvt_f16_kernel(const float* __restrict__ in,
                               _Float16* __restrict__ out, int n) {
  int i = blockIdx.x * 256 + threadIdx.x;
  if (i < n) out[i] = (_Float16)in[i];
}

__global__ void proj_tr_kernel(const float* __restrict__ proj,
                               _Float16* __restrict__ projT) {
  using namespace cfg;
  int i = blockIdx.x * 256 + threadIdx.x;
  if (i >= C * DOUT) return;
  int c = i / DOUT, d = i % DOUT;
  projT[d * C + c] = (_Float16)proj[i];
}

// ---------------------------------------------------------------------------
// LayerNorm over C=768 -> f16; zero padded rows. One 256-thread block per row.
// ---------------------------------------------------------------------------
__global__ void ln_kernel(const float* __restrict__ x, const float* __restrict__ g,
                          const float* __restrict__ b, _Float16* __restrict__ out) {
  using namespace cfg;
  int row = blockIdx.x;          // T*SP rows
  int s = row % SP;
  int tid = threadIdx.x;
  _Float16* orow = out + (size_t)row * C;
  if (s >= S) {
    for (int c = tid; c < C; c += 256) orow[c] = (_Float16)0.0f;
    return;
  }
  const float* xr = x + (size_t)row * C;
  __shared__ float red[256];
  float lsum = 0.0f;
  for (int c = tid; c < C; c += 256) lsum += xr[c];
  red[tid] = lsum; __syncthreads();
  for (int o = 128; o; o >>= 1) { if (tid < o) red[tid] += red[tid + o]; __syncthreads(); }
  float mean = red[0] / C;
  __syncthreads();
  float lv = 0.0f;
  for (int c = tid; c < C; c += 256) { float d = xr[c] - mean; lv += d * d; }
  red[tid] = lv; __syncthreads();
  for (int o = 128; o; o >>= 1) { if (tid < o) red[tid] += red[tid + o]; __syncthreads(); }
  float rstd = rsqrtf(red[0] / C + 1e-5f);
  for (int c = tid; c < C; c += 256)
    orow[c] = (_Float16)((xr[c] - mean) * rstd * g[c] + b[c]);
}

// ln_post over the Q sos tokens -> pbuf [T*Q, C] f16
__global__ void lnp_kernel(const float* __restrict__ x, const float* __restrict__ g,
                           const float* __restrict__ b, _Float16* __restrict__ out) {
  using namespace cfg;
  int rid = blockIdx.x;          // t*Q + q
  int t = rid / Q, q = rid % Q;
  int tid = threadIdx.x;
  const float* xr = x + ((size_t)t * SP + q) * C;
  _Float16* orow = out + (size_t)rid * C;
  __shared__ float red[256];
  float lsum = 0.0f;
  for (int c = tid; c < C; c += 256) lsum += xr[c];
  red[tid] = lsum; __syncthreads();
  for (int o = 128; o; o >>= 1) { if (tid < o) red[tid] += red[tid + o]; __syncthreads(); }
  float mean = red[0] / C;
  __syncthreads();
  float lv = 0.0f;
  for (int c = tid; c < C; c += 256) { float d = xr[c] - mean; lv += d * d; }
  red[tid] = lv; __syncthreads();
  for (int o = 128; o; o >>= 1) { if (tid < o) red[tid] += red[tid + o]; __syncthreads(); }
  float rstd = rsqrtf(red[0] / C + 1e-5f);
  for (int c = tid; c < C; c += 256)
    orow[c] = (_Float16)((xr[c] - mean) * rstd * g[c] + b[c]);
}

// ---------------------------------------------------------------------------
// Generic WMMA GEMM: out[M,N] = act(A[M,K] @ B[N,K]^T + bias) + resid
// A,B f16 row-major.  Block = 4 waves; each wave produces a 32(M) x 64(N)
// tile: 2 A frags x 4 B frags = 8 WMMAs per K step.  Each B fragment load is
// interleaved with the pair of WMMAs consuming it, keeping <= ~100 VGPRs live
// (no spills) while leaving the scheduler free to clause/hoist the loads;
// remaining latency is hidden by multi-wave occupancy.  OOB N columns are
// clamped at load (results discarded at store) -> branch-free inner loop.
// act: 0 = none, 1 = QuickGELU. Output to f32 and/or f16 buffers.
// ---------------------------------------------------------------------------
__global__ void gemm_f16_kernel(const _Float16* __restrict__ A,
                                const _Float16* __restrict__ B,
                                const float* __restrict__ bias,
                                const float* resid,       // may alias outF
                                float* outF, _Float16* outH,
                                int Mm, int Nn, int Kk, int act) {
  int wave = threadIdx.x >> 5;
  int lane = threadIdx.x & 31;
  int mbase = blockIdx.y * 128 + wave * 32;
  int nbase = blockIdx.x * 64;
  if (mbase >= Mm) return;
  int l15 = lane & 15;
  int khalf = (lane >> 4) * 16;     // which 16-element K chunk this lane holds

  const _Float16* pa0 = A + (size_t)(mbase + l15) * Kk + khalf;
  const _Float16* pa1 = pa0 + (size_t)16 * Kk;
  int c0 = nbase + l15;       if (c0 >= Nn) c0 = Nn - 1;   // clamp: loads stay
  int c1 = nbase + 16 + l15;  if (c1 >= Nn) c1 = Nn - 1;   // in-bounds, results
  int c2 = nbase + 32 + l15;  if (c2 >= Nn) c2 = Nn - 1;   // discarded at store
  int c3 = nbase + 48 + l15;  if (c3 >= Nn) c3 = Nn - 1;
  const _Float16* pb0 = B + (size_t)c0 * Kk + khalf;
  const _Float16* pb1 = B + (size_t)c1 * Kk + khalf;
  const _Float16* pb2 = B + (size_t)c2 * Kk + khalf;
  const _Float16* pb3 = B + (size_t)c3 * Kk + khalf;

  v8f acc[8];
#pragma unroll
  for (int i = 0; i < 8; i++) acc[i] = v8f{};

  for (int k0 = 0; k0 < Kk; k0 += 32) {
    v16h a0 = *(const v16h*)(pa0 + k0);
    v16h a1 = *(const v16h*)(pa1 + k0);
    v16h b0 = *(const v16h*)(pb0 + k0);
    acc[0] = __builtin_amdgcn_wmma_f32_16x16x32_f16(false, a0, false, b0, (short)0, acc[0], false, false);
    acc[4] = __builtin_amdgcn_wmma_f32_16x16x32_f16(false, a1, false, b0, (short)0, acc[4], false, false);
    v16h b1 = *(const v16h*)(pb1 + k0);
    acc[1] = __builtin_amdgcn_wmma_f32_16x16x32_f16(false, a0, false, b1, (short)0, acc[1], false, false);
    acc[5] = __builtin_amdgcn_wmma_f32_16x16x32_f16(false, a1, false, b1, (short)0, acc[5], false, false);
    v16h b2 = *(const v16h*)(pb2 + k0);
    acc[2] = __builtin_amdgcn_wmma_f32_16x16x32_f16(false, a0, false, b2, (short)0, acc[2], false, false);
    acc[6] = __builtin_amdgcn_wmma_f32_16x16x32_f16(false, a1, false, b2, (short)0, acc[6], false, false);
    v16h b3 = *(const v16h*)(pb3 + k0);
    acc[3] = __builtin_amdgcn_wmma_f32_16x16x32_f16(false, a0, false, b3, (short)0, acc[3], false, false);
    acc[7] = __builtin_amdgcn_wmma_f32_16x16x32_f16(false, a1, false, b3, (short)0, acc[7], false, false);
  }

  int hi = (lane >> 4) * 8;
#pragma unroll
  for (int ai = 0; ai < 2; ai++) {
#pragma unroll
    for (int j = 0; j < 4; j++) {
      int ccol = nbase + j * 16 + l15;
      if (ccol >= Nn) continue;
      float bv = bias ? bias[ccol] : 0.0f;
#pragma unroll
      for (int r = 0; r < 8; r++) {
        int row = mbase + ai * 16 + r + hi;
        float v = acc[ai * 4 + j][r] + bv;
        if (act) v = v / (1.0f + __expf(-1.702f * v));   // QuickGELU
        size_t idx = (size_t)row * Nn + ccol;
        if (resid) v += resid[idx];
        if (outF) outF[idx] = v;
        if (outH) outH[idx] = (_Float16)v;
      }
    }
  }
}

// ---------------------------------------------------------------------------
// Split qkv f32 [T,SP,3C] -> qh,kh [T,NH,SP,HD] f16 and vT [T,NH,HD,SP] f16.
// ---------------------------------------------------------------------------
__global__ void split_qkv_kernel(const float* __restrict__ qkv,
                                 _Float16* __restrict__ qh,
                                 _Float16* __restrict__ kh,
                                 _Float16* __restrict__ vt) {
  using namespace cfg;
  int idx = blockIdx.x * 256 + threadIdx.x;
  if (idx >= M * C) return;
  int c = idx % C;
  int s = (idx / C) % SP;
  int t = idx / (C * SP);
  int h = c / HD, d = c % HD;
  const float* base = qkv + ((size_t)t * SP + s) * 3 * C;
  size_t th = (size_t)t * NH + h;
  qh[(th * SP + s) * HD + d] = (_Float16)base[c];
  kh[(th * SP + s) * HD + d] = (_Float16)base[C + c];
  vt[(th * HD + d) * SP + s] = (_Float16)base[2 * C + c];
}

// ---------------------------------------------------------------------------
// Attention scores: sc[t,h,s,u] = scale * q·k + bias(t,h,s,u).
// One wave per 16x16 tile; K=64 -> two 16x16x32 WMMAs.
// ---------------------------------------------------------------------------
__global__ void attn_score_kernel(const _Float16* __restrict__ qh,
                                  const _Float16* __restrict__ kh,
                                  const float* __restrict__ pooled,
                                  float* __restrict__ sc) {
  using namespace cfg;
  int lane = threadIdx.x;
  int mt = blockIdx.x / (SP / 16);
  int nt = blockIdx.x % (SP / 16);
  int h = blockIdx.y, t = blockIdx.z;
  size_t th = (size_t)t * NH + h;
  const _Float16* q = qh + th * SP * HD;
  const _Float16* k = kh + th * SP * HD;
  int khalf = (lane >> 4) * 16;
  int qrow = mt * 16 + (lane & 15);
  int krow = nt * 16 + (lane & 15);

  v8f acc = v8f{};
#pragma unroll
  for (int k0 = 0; k0 < HD; k0 += 32) {
    v16h a = *(const v16h*)(q + (size_t)qrow * HD + k0 + khalf);
    v16h b = *(const v16h*)(k + (size_t)krow * HD + k0 + khalf);
    acc = __builtin_amdgcn_wmma_f32_16x16x32_f16(false, a, false, b,
                                                 (short)0, acc, false, false);
  }

  int hi = (lane >> 4) * 8;
  int ucol = nt * 16 + (lane & 15);
  float* out = sc + th * SP * SP;
#pragma unroll
  for (int r = 0; r < 8; r++) {
    int srow = mt * 16 + r + hi;
    float bias;
    if (ucol >= S)       bias = -1e30f;                       // padded column
    else if (ucol < Q)   bias = (srow == ucol) ? 0.0f : -100.0f;
    else if (ucol == Q)  bias = (srow < Q) ? -100.0f : 0.0f;
    else if (srow < Q)   bias = pooled[((size_t)t * Q + srow) * L + (ucol - Q - 1)];
    else                 bias = 0.0f;
    out[(size_t)srow * SP + ucol] = acc[r] * SCALE + bias;
  }
}

// Softmax over u (SP cols, padded cols already -1e30) -> f16 probs.
__global__ void softmax_kernel(const float* __restrict__ sc,
                               _Float16* __restrict__ ah) {
  using namespace cfg;
  int s = blockIdx.x, h = blockIdx.y, t = blockIdx.z;
  size_t roff = (((size_t)t * NH + h) * SP + s) * SP;
  const float* row = sc + roff;
  _Float16* orow = ah + roff;
  int tid = threadIdx.x;
  if (s >= S) {                      // padded row: probs = 0
    for (int u = tid; u < SP; u += 128) orow[u] = (_Float16)0.0f;
    return;
  }
  __shared__ float red[128];
  __shared__ float buf[cfg::SP];
  float m = -1e38f;
  for (int u = tid; u < SP; u += 128) { float v = row[u]; buf[u] = v; m = fmaxf(m, v); }
  red[tid] = m; __syncthreads();
  for (int o = 64; o; o >>= 1) { if (tid < o) red[tid] = fmaxf(red[tid], red[tid + o]); __syncthreads(); }
  m = red[0]; __syncthreads();
  float sum = 0.0f;
  for (int u = tid; u < SP; u += 128) { float e = __expf(buf[u] - m); buf[u] = e; sum += e; }
  red[tid] = sum; __syncthreads();
  for (int o = 64; o; o >>= 1) { if (tid < o) red[tid] += red[tid + o]; __syncthreads(); }
  float inv = 1.0f / red[0];
  for (int u = tid; u < SP; u += 128) orow[u] = (_Float16)(buf[u] * inv);
}

// ---------------------------------------------------------------------------
// o = A(probs) @ V per (t,h). Wave: 16 rows x all 64 head dims; K loop SP/32.
// Writes concat-head f16 activations [T,SP,C] ready for the out-proj GEMM.
// ---------------------------------------------------------------------------
__global__ void attn_av_kernel(const _Float16* __restrict__ ah,
                               const _Float16* __restrict__ vt,
                               _Float16* __restrict__ oh) {
  using namespace cfg;
  int wave = threadIdx.x >> 5;
  int lane = threadIdx.x & 31;
  int mt = blockIdx.x * 4 + wave;
  int h = blockIdx.y, t = blockIdx.z;
  size_t th = (size_t)t * NH + h;
  const _Float16* a = ah + th * SP * SP;
  const _Float16* v = vt + th * HD * SP;
  int khalf = (lane >> 4) * 16;
  int arow = mt * 16 + (lane & 15);

  v8f acc[4];
  acc[0] = v8f{}; acc[1] = v8f{}; acc[2] = v8f{}; acc[3] = v8f{};
  for (int k0 = 0; k0 < SP; k0 += 32) {
    v16h af = *(const v16h*)(a + (size_t)arow * SP + k0 + khalf);
#pragma unroll
    for (int d = 0; d < 4; d++) {
      v16h bf = *(const v16h*)(v + (size_t)(d * 16 + (lane & 15)) * SP + k0 + khalf);
      acc[d] = __builtin_amdgcn_wmma_f32_16x16x32_f16(false, af, false, bf,
                                                      (short)0, acc[d], false, false);
    }
  }
  int hi = (lane >> 4) * 8;
#pragma unroll
  for (int d = 0; d < 4; d++) {
    int dcol = d * 16 + (lane & 15);
#pragma unroll
    for (int r = 0; r < 8; r++) {
      int srow = mt * 16 + r + hi;
      oh[((size_t)t * SP + srow) * C + h * HD + dcol] = (_Float16)acc[d][r];
    }
  }
}

// Row L2-normalize f32 -> f16 (projected sos tokens).
__global__ void rownorm_kernel(const float* __restrict__ in,
                               _Float16* __restrict__ out, int ncols) {
  int row = blockIdx.x, tid = threadIdx.x;
  const float* r = in + (size_t)row * ncols;
  __shared__ float red[128];
  float ss = 0.0f;
  for (int c = tid; c < ncols; c += 128) { float v = r[c]; ss += v * v; }
  red[tid] = ss; __syncthreads();
  for (int o = 64; o; o >>= 1) { if (tid < o) red[tid] += red[tid + o]; __syncthreads(); }
  float rn = rsqrtf(red[0]);
  for (int c = tid; c < ncols; c += 128)
    out[(size_t)row * ncols + c] = (_Float16)(r[c] * rn);
}

// Normalize text rows and fold in exp(logit_scale).
__global__ void textnorm_kernel(const float* __restrict__ txt,
                                const float* __restrict__ ls,
                                _Float16* __restrict__ out) {
  using namespace cfg;
  int row = blockIdx.x, tid = threadIdx.x;
  const float* r = txt + (size_t)row * DOUT;
  __shared__ float red[128];
  float ss = 0.0f;
  for (int c = tid; c < DOUT; c += 128) { float v = r[c]; ss += v * v; }
  red[tid] = ss; __syncthreads();
  for (int o = 64; o; o >>= 1) { if (tid < o) red[tid] += red[tid + o]; __syncthreads(); }
  float scl = rsqrtf(red[0]) * __expf(ls[0]);
  for (int c = tid; c < DOUT; c += 128)
    out[(size_t)row * DOUT + c] = (_Float16)(r[c] * scl);
}

// ---------------------------------------------------------------------------
extern "C" void kernel_launch(void* const* d_in, const int* in_sizes, int n_in,
                              void* d_out, int out_size, void* d_ws, size_t ws_size,
                              hipStream_t stream) {
  using namespace cfg;
  const float* cls   = (const float*)d_in[0];
  const float* pix   = (const float*)d_in[1];
  const float* masks = (const float*)d_in[2];
  const float* txt   = (const float*)d_in[3];
  const float* ls    = (const float*)d_in[4];
  const float* ln1g  = (const float*)d_in[5];
  const float* ln1b  = (const float*)d_in[6];
  const float* wqkv  = (const float*)d_in[7];
  const float* bqkv  = (const float*)d_in[8];
  const float* wout  = (const float*)d_in[9];
  const float* bout  = (const float*)d_in[10];
  const float* ln2g  = (const float*)d_in[11];
  const float* ln2b  = (const float*)d_in[12];
  const float* wfc1  = (const float*)d_in[13];
  const float* bfc1  = (const float*)d_in[14];
  const float* wfc2  = (const float*)d_in[15];
  const float* bfc2  = (const float*)d_in[16];
  const float* lnpg  = (const float*)d_in[17];
  const float* lnpb  = (const float*)d_in[18];
  const float* proj  = (const float*)d_in[19];

  size_t off = 0;
  auto take = [&](size_t bytes) -> char* {
    char* p = (char*)d_ws + off;
    off = (off + bytes + 255) & ~(size_t)255;
    return p;
  };
  float*    pooled = (float*)   take((size_t)T * Q * L * 4);
  float*    x      = (float*)   take((size_t)M * C * 4);
  _Float16* hnh    = (_Float16*)take((size_t)M * C * 2);
  float*    qkv    = (float*)   take((size_t)M * 3 * C * 4);
  _Float16* qh     = (_Float16*)take((size_t)T * NH * SP * HD * 2);
  _Float16* kh     = (_Float16*)take((size_t)T * NH * SP * HD * 2);
  _Float16* vt     = (_Float16*)take((size_t)T * NH * SP * HD * 2);
  float*    sc     = (float*)   take((size_t)T * NH * SP * SP * 4);
  _Float16* ahp    = (_Float16*)take((size_t)T * NH * SP * SP * 2);
  _Float16* oh     = (_Float16*)take((size_t)M * C * 2);
  _Float16* hid    = (_Float16*)take((size_t)M * 4 * C * 2);
  _Float16* wqkvh  = (_Float16*)take((size_t)NL * 3 * C * C * 2);
  _Float16* wouth  = (_Float16*)take((size_t)NL * C * C * 2);
  _Float16* wfc1h  = (_Float16*)take((size_t)NL * 4 * C * C * 2);
  _Float16* wfc2h  = (_Float16*)take((size_t)NL * 4 * C * C * 2);
  _Float16* projTh = (_Float16*)take((size_t)DOUT * C * 2);
  _Float16* pbuf   = (_Float16*)take((size_t)T * Q * C * 2);
  float*    outt   = (float*)   take((size_t)T * Q * DOUT * 4);
  _Float16* nsh    = (_Float16*)take((size_t)T * Q * DOUT * 2);
  _Float16* txth   = (_Float16*)take((size_t)NTXT * DOUT * 2);

  // Stage 0: bias pooling, token assembly, f16 weight prep, text prep.
  pool_bias_kernel<<<T * Q * L, 256, 0, stream>>>(masks, pooled);
  assemble_x_kernel<<<(M * C + 255) / 256, 256, 0, stream>>>(cls, pix, x);
  cvt_f16_kernel<<<(NL * 3 * C * C + 255) / 256, 256, 0, stream>>>(wqkv, wqkvh, NL * 3 * C * C);
  cvt_f16_kernel<<<(NL * C * C + 255) / 256, 256, 0, stream>>>(wout, wouth, NL * C * C);
  cvt_f16_kernel<<<(NL * 4 * C * C + 255) / 256, 256, 0, stream>>>(wfc1, wfc1h, NL * 4 * C * C);
  cvt_f16_kernel<<<(NL * 4 * C * C + 255) / 256, 256, 0, stream>>>(wfc2, wfc2h, NL * 4 * C * C);
  proj_tr_kernel<<<(C * DOUT + 255) / 256, 256, 0, stream>>>(proj, projTh);
  textnorm_kernel<<<NTXT, 128, 0, stream>>>(txt, ls, txth);

  // 3 transformer blocks.
  for (int i = 0; i < NL; i++) {
    ln_kernel<<<M, 256, 0, stream>>>(x, ln1g + i * C, ln1b + i * C, hnh);
    gemm_f16_kernel<<<dim3(3 * C / 64, (M + 127) / 128), 128, 0, stream>>>(
        hnh, wqkvh + (size_t)i * 3 * C * C, bqkv + i * 3 * C,
        nullptr, qkv, nullptr, M, 3 * C, C, 0);
    split_qkv_kernel<<<(M * C + 255) / 256, 256, 0, stream>>>(qkv, qh, kh, vt);
    attn_score_kernel<<<dim3((SP / 16) * (SP / 16), NH, T), 32, 0, stream>>>(qh, kh, pooled, sc);
    softmax_kernel<<<dim3(SP, NH, T), 128, 0, stream>>>(sc, ahp);
    attn_av_kernel<<<dim3(SP / 64, NH, T), 128, 0, stream>>>(ahp, vt, oh);
    gemm_f16_kernel<<<dim3(C / 64, (M + 127) / 128), 128, 0, stream>>>(
        oh, wouth + (size_t)i * C * C, bout + i * C, x, x, nullptr, M, C, C, 0);
    ln_kernel<<<M, 256, 0, stream>>>(x, ln2g + i * C, ln2b + i * C, hnh);
    gemm_f16_kernel<<<dim3(4 * C / 64, (M + 127) / 128), 128, 0, stream>>>(
        hnh, wfc1h + (size_t)i * 4 * C * C, bfc1 + i * 4 * C,
        nullptr, nullptr, hid, M, 4 * C, C, 1);
    gemm_f16_kernel<<<dim3(C / 64, (M + 127) / 128), 128, 0, stream>>>(
        hid, wfc2h + (size_t)i * 4 * C * C, bfc2 + i * C, x, x, nullptr, M, C, 4 * C, 0);
  }

  // Head: ln_post -> proj -> normalize -> logits.
  lnp_kernel<<<T * Q, 256, 0, stream>>>(x, lnpg, lnpb, pbuf);
  gemm_f16_kernel<<<dim3(DOUT / 64, (T * Q + 127) / 128), 128, 0, stream>>>(
      pbuf, projTh, nullptr, nullptr, outt, nullptr, T * Q, DOUT, C, 0);
  rownorm_kernel<<<T * Q, 128, 0, stream>>>(outt, nsh, DOUT);
  gemm_f16_kernel<<<dim3(1, (T * Q + 127) / 128), 128, 0, stream>>>(
      nsh, txth, nullptr, nullptr, (float*)d_out, nullptr, T * Q, NTXT, DOUT, 0);
}
